// YOLO_16475494547852
// MI455X (gfx1250) — compile-verified, matrix operand order checked by reading, not run on hardware
//
#include <hip/hip_runtime.h>
#include <hip/hip_bf16.h>

// ---------------------------------------------------------------------------
// YOLO detection head for MI455X (gfx1250, wave32, WMMA).
// Pipeline: convert weights+features to f16 -> WMMA GEMM (255x C per pixel)
// -> score histogram top-k selection -> rank sort -> greedy NMS -> outputs.
// ---------------------------------------------------------------------------

typedef __attribute__((ext_vector_type(16))) _Float16 v16h;
typedef __attribute__((ext_vector_type(8)))  float    v8f;

union V16 { v16h v; _Float16 h[16]; };

#define NC       80
#define BPC      3
#define NPAD     256     // 255 head outputs padded to 256 (16 N-tiles)
#define TOPK     1000
#define KCAND    3000
#define NBINS    4096
#define CONF_THR 0.3f
#define NMS_THR  0.5f

__device__ __forceinline__ float sigm(float x) { return 1.0f / (1.0f + expf(-x)); }

// ---------------------------------------------------------------------------
// zero int32 region
// ---------------------------------------------------------------------------
__global__ void k_zero_i32(int* __restrict__ p, int n) {
    int t = blockIdx.x * blockDim.x + threadIdx.x;
    if (t < n) p[t] = 0;
}

// ---------------------------------------------------------------------------
// Pack obj/cls/reg weights into one [256][C] f16 matrix + bias vector.
// Row map: 0..2 obj(b), 3..242 cls(b*80+c), 243..254 reg(b*4+j), 255 zero.
// ---------------------------------------------------------------------------
__global__ void k_cvt_weights(const float* __restrict__ wobj, const float* __restrict__ bobj,
                              const float* __restrict__ wcls, const float* __restrict__ bcls,
                              const float* __restrict__ wreg, const float* __restrict__ breg,
                              _Float16* __restrict__ wcat, float* __restrict__ bcat, int C) {
    int t = blockIdx.x * blockDim.x + threadIdx.x;
    int total = NPAD * C;
    if (t >= total) return;
    int r = t / C, c = t - r * C;
    float v = 0.0f, bv = 0.0f;
    if (r < BPC)            { v = wobj[r * C + c];            bv = bobj[r]; }
    else if (r < 3 + BPC*NC){ v = wcls[(r - 3) * C + c];      bv = bcls[r - 3]; }
    else if (r < 255)       { v = wreg[(r - 243) * C + c];    bv = breg[r - 243]; }
    wcat[t] = (_Float16)v;
    if (c == 0) bcat[r] = bv;
}

// ---------------------------------------------------------------------------
// Transpose + convert features: [C][HW] f32 -> [HW][C] f16 (K contiguous per
// pixel so WMMA A fragments are two b128 loads per lane).
// ---------------------------------------------------------------------------
__global__ void k_cvt_feat(const float* __restrict__ feat, _Float16* __restrict__ featT,
                           int C, int HW) {
    int t = blockIdx.x * blockDim.x + threadIdx.x;
    if (t >= C * HW) return;
    int c = t / HW, m = t - c * HW;
    featT[(size_t)m * C + c] = (_Float16)feat[t];
}

// ---------------------------------------------------------------------------
// WMMA GEMM: out[M][256] = featT[M][C] * wcat[256][C]^T + bias.
// One wave per 16x16 tile, K stepped by 32 with v_wmma_f32_16x16x32_f16.
// A layout (16-bit, 16x32): lanes 0-15 hold M=lane, K 0..7 / 16..23;
// lanes 16-31 hold M=lane-16, K 8..15 / 24..31 -> two contiguous b128 loads.
// B mirrors A over the N dimension (wcat rows are N, K contiguous).
// ---------------------------------------------------------------------------
__global__ __launch_bounds__(32)
void k_gemm_head(const _Float16* __restrict__ featT, const _Float16* __restrict__ wcat,
                 const float* __restrict__ bcat, float* __restrict__ outg, int C) {
    const int lane = threadIdx.x;
    const int nTile = blockIdx.x & 15;
    const int mTile = blockIdx.x >> 4;
    const int m0 = mTile * 16;
    const int n0 = nTile * 16;
    const int half = (lane >> 4) & 1;
    const int l15  = lane & 15;

    const _Float16* pa = featT + (size_t)(m0 + l15) * C + half * 8;
    const _Float16* pb = wcat  + (size_t)(n0 + l15) * C + half * 8;

    v8f acc = {};
    const int ksteps = C >> 5;
    for (int ks = 0; ks < ksteps; ++ks) {
        V16 a, b;
        *(uint4*)&a.h[0] = *(const uint4*)(pa);
        *(uint4*)&a.h[8] = *(const uint4*)(pa + 16);
        *(uint4*)&b.h[0] = *(const uint4*)(pb);
        *(uint4*)&b.h[8] = *(const uint4*)(pb + 16);
        __builtin_prefetch((const void*)(pa + 32), 0, 1);   // global_prefetch_b8
        __builtin_prefetch((const void*)(pb + 32), 0, 1);
        acc = __builtin_amdgcn_wmma_f32_16x16x32_f16(
                  /*neg_a=*/false, a.v, /*neg_b=*/false, b.v,
                  /*c_mod=*/(short)0, acc, /*reuse_a=*/false, /*reuse_b=*/false);
        pa += 32; pb += 32;
    }

    const float bias = bcat[n0 + l15];
    // D layout: VGPR r -> row m0 + r + 8*half, col n0 + l15
    float* po = outg + (size_t)(m0 + half * 8) * NPAD + n0 + l15;
#pragma unroll
    for (int r = 0; r < 8; ++r)
        po[(size_t)r * NPAD] = acc[r] + bias;
}

// ---------------------------------------------------------------------------
// Score histogram (4096 bins over [0,1)) for top-k threshold selection.
// ---------------------------------------------------------------------------
__global__ void k_hist(const float* __restrict__ outg, unsigned int* __restrict__ hist, int M) {
    int t = blockIdx.x * blockDim.x + threadIdx.x;
    if (t >= M * BPC * NC) return;
    int m = t / (BPC * NC), r = t - m * (BPC * NC);
    int b = r / NC, c = r - b * NC;
    float obj = outg[m * NPAD + b];
    float cls = outg[m * NPAD + 3 + b * NC + c];
    float s = sqrtf(sigm(obj) * sigm(cls));
    int bin = (int)(s * (float)NBINS);
    bin = bin < 0 ? 0 : (bin > NBINS - 1 ? NBINS - 1 : bin);
    atomicAdd(&hist[bin], 1u);
}

__global__ void k_find_thr(const unsigned int* __restrict__ hist, float* __restrict__ thr) {
    if (threadIdx.x == 0 && blockIdx.x == 0) {
        unsigned int acc = 0;
        int b = NBINS - 1;
        for (; b >= 0; --b) { acc += hist[b]; if (acc >= (unsigned)TOPK) break; }
        if (b < 0) b = 0;
        *thr = (float)b / (float)NBINS;
    }
}

// ---------------------------------------------------------------------------
// Collect candidates above threshold (cap 1000/level) + decode boxes.
// ---------------------------------------------------------------------------
__global__ void k_collect(const float* __restrict__ outg, const float* __restrict__ thr,
                          int* __restrict__ cnt,
                          float* __restrict__ sc_g, int* __restrict__ lab_g,
                          float* __restrict__ box_g,
                          int slot0, int M, int S, float stride,
                          float aw0, float ah0, float aw1, float ah1, float aw2, float ah2) {
    int t = blockIdx.x * blockDim.x + threadIdx.x;
    if (t >= M * BPC * NC) return;
    int m = t / (BPC * NC), r = t - m * (BPC * NC);
    int b = r / NC, c = r - b * NC;
    float obj = outg[m * NPAD + b];
    float cls = outg[m * NPAD + 3 + b * NC + c];
    float s = sqrtf(sigm(obj) * sigm(cls));
    if (s < *thr) return;
    int slot = atomicAdd(cnt, 1);
    if (slot >= TOPK) return;
    int idx = slot0 + slot;

    float tx = outg[m * NPAD + 243 + b * 4 + 0];
    float ty = outg[m * NPAD + 243 + b * 4 + 1];
    float tw = outg[m * NPAD + 243 + b * 4 + 2];
    float th = outg[m * NPAD + 243 + b * 4 + 3];
    float gx = (float)(m % S), gy = (float)(m / S);
    float aw = (b == 0) ? aw0 : (b == 1 ? aw1 : aw2);
    float ah = (b == 0) ? ah0 : (b == 1 ? ah1 : ah2);
    float cx = (sigm(tx) + gx) * stride;
    float cy = (sigm(ty) + gy) * stride;
    float w  = expf(tw) * aw;
    float h  = expf(th) * ah;

    sc_g[idx]  = s;
    lab_g[idx] = c;
    box_g[idx * 4 + 0] = cx - 0.5f * w;
    box_g[idx * 4 + 1] = cy - 0.5f * h;
    box_g[idx * 4 + 2] = cx + 0.5f * w;
    box_g[idx * 4 + 3] = cy + 0.5f * h;
}

// ---------------------------------------------------------------------------
// Stable descending rank sort of 3000 scores (O(K^2) in LDS).
// ---------------------------------------------------------------------------
__global__ __launch_bounds__(1024)
void k_rank(const float* __restrict__ sc_g, int* __restrict__ order_g) {
    __shared__ float s[KCAND];
    for (int i = threadIdx.x; i < KCAND; i += 1024) s[i] = sc_g[i];
    __syncthreads();
    for (int i = threadIdx.x; i < KCAND; i += 1024) {
        float si = s[i];
        int rk = 0;
        for (int j = 0; j < KCAND; ++j) {
            float sj = s[j];
            rk += (sj > si) || ((sj == si) && (j < i));
        }
        order_g[rk] = i;
    }
}

// ---------------------------------------------------------------------------
// Greedy per-class NMS (single block) + final masked outputs.
// Faithful to the reference's cxcywh->xyxy re-conversion quirk on xyxy boxes.
// ---------------------------------------------------------------------------
__global__ __launch_bounds__(1024)
void k_nms_out(const float* __restrict__ sc_g, const int* __restrict__ lab_g,
               const float* __restrict__ box_g, const int* __restrict__ order_g,
               float* __restrict__ out) {
    __shared__ float sx1[KCAND], sy1[KCAND], sx2[KCAND], sy2[KCAND];
    __shared__ short slab[KCAND];
    __shared__ unsigned char skeep[KCAND];
    __shared__ unsigned char korig[KCAND];
    const int tid = threadIdx.x;

    for (int k = tid; k < KCAND; k += 1024) {
        int i = order_g[k];
        float b0 = box_g[i * 4 + 0], b1 = box_g[i * 4 + 1];
        float b2 = box_g[i * 4 + 2], b3 = box_g[i * 4 + 3];
        sx1[k] = b0 - 0.5f * b2;   // quirk: treat xyxy as cxcywh
        sy1[k] = b1 - 0.5f * b3;
        sx2[k] = b0 + 0.5f * b2;
        sy2[k] = b1 + 0.5f * b3;
        slab[k]  = (short)lab_g[i];
        skeep[k] = (sc_g[i] > CONF_THR) ? 1 : 0;
    }
    __syncthreads();

    for (int k = 0; k < KCAND; ++k) {
        __syncthreads();
        if (!skeep[k]) continue;            // uniform branch (same LDS value)
        float x1 = sx1[k], y1 = sy1[k], x2 = sx2[k], y2 = sy2[k];
        float ak = (x2 - x1) * (y2 - y1);
        short lk = slab[k];
        for (int j = k + 1 + tid; j < KCAND; j += 1024) {
            if (slab[j] != lk) continue;
            float xx1 = fmaxf(x1, sx1[j]), yy1 = fmaxf(y1, sy1[j]);
            float xx2 = fminf(x2, sx2[j]), yy2 = fminf(y2, sy2[j]);
            float w = fmaxf(1e-10f, xx2 - xx1);
            float h = fmaxf(1e-10f, yy2 - yy1);
            float inter = w * h;
            float aj = (sx2[j] - sx1[j]) * (sy2[j] - sy1[j]);
            float iou = inter / (ak + aj - inter + 1e-14f);
            if (iou > NMS_THR) skeep[j] = 0;
        }
    }
    __syncthreads();
    for (int k = tid; k < KCAND; k += 1024) korig[order_g[k]] = skeep[k];
    __syncthreads();

    // outputs: boxes[3000][4], scores[3000], labels[3000], keep[3000]
    for (int i = tid; i < KCAND; i += 1024) {
        float m = korig[i] ? 1.0f : 0.0f;
        out[i * 4 + 0] = box_g[i * 4 + 0] * m;
        out[i * 4 + 1] = box_g[i * 4 + 1] * m;
        out[i * 4 + 2] = box_g[i * 4 + 2] * m;
        out[i * 4 + 3] = box_g[i * 4 + 3] * m;
        out[4 * KCAND + i]          = sc_g[i] * m;
        out[5 * KCAND + i]          = (float)lab_g[i];
        out[6 * KCAND + i]          = m;
    }
}

// ---------------------------------------------------------------------------
// Host launcher
// ---------------------------------------------------------------------------
extern "C" void kernel_launch(void* const* d_in, const int* in_sizes, int n_in,
                              void* d_out, int out_size, void* d_ws, size_t ws_size,
                              hipStream_t stream) {
    (void)in_sizes; (void)n_in; (void)out_size; (void)ws_size;
    static const int   C[3]   = {256, 512, 1024};
    static const int   S[3]   = {160, 80, 40};
    static const int   HW[3]  = {25600, 6400, 1600};
    static const float STR[3] = {8.0f, 16.0f, 32.0f};
    static const float ANC[3][6] = {
        {10, 13, 16, 30, 33, 23},
        {30, 61, 62, 45, 59, 119},
        {116, 90, 156, 198, 373, 326}};

    char* ws = (char*)d_ws;
    size_t off = 0;
    auto alloc = [&](size_t bytes) {
        size_t o = off;
        off = (off + bytes + 255) & ~(size_t)255;
        return o;
    };
    size_t o_feat[3], o_wcat[3], o_bcat[3], o_outg[3];
    for (int l = 0; l < 3; ++l) o_feat[l] = alloc((size_t)HW[l] * C[l] * sizeof(_Float16));
    for (int l = 0; l < 3; ++l) o_wcat[l] = alloc((size_t)NPAD * C[l] * sizeof(_Float16));
    for (int l = 0; l < 3; ++l) o_bcat[l] = alloc(NPAD * sizeof(float));
    for (int l = 0; l < 3; ++l) o_outg[l] = alloc((size_t)HW[l] * NPAD * sizeof(float));
    size_t o_small = off;
    size_t o_hist = alloc(3 * NBINS * sizeof(unsigned int));
    size_t o_cnt  = alloc(64);
    size_t o_thr  = alloc(64);
    size_t o_sc   = alloc(KCAND * sizeof(float));
    size_t o_lab  = alloc(KCAND * sizeof(int));
    size_t o_box  = alloc(KCAND * 4 * sizeof(float));
    size_t o_ord  = alloc(KCAND * sizeof(int));
    int zn = (int)((off - o_small) / 4);

    k_zero_i32<<<(zn + 255) / 256, 256, 0, stream>>>((int*)(ws + o_small), zn);

    for (int l = 0; l < 3; ++l) {
        const float* feat = (const float*)d_in[7 * l + 0];
        const float* wobj = (const float*)d_in[7 * l + 1];
        const float* bobj = (const float*)d_in[7 * l + 2];
        const float* wcls = (const float*)d_in[7 * l + 3];
        const float* bcls = (const float*)d_in[7 * l + 4];
        const float* wreg = (const float*)d_in[7 * l + 5];
        const float* breg = (const float*)d_in[7 * l + 6];

        _Float16* wcat = (_Float16*)(ws + o_wcat[l]);
        float*    bcat = (float*)(ws + o_bcat[l]);
        _Float16* fT   = (_Float16*)(ws + o_feat[l]);
        float*    outg = (float*)(ws + o_outg[l]);

        int wtot = NPAD * C[l];
        k_cvt_weights<<<(wtot + 255) / 256, 256, 0, stream>>>(
            wobj, bobj, wcls, bcls, wreg, breg, wcat, bcat, C[l]);

        int ftot = C[l] * HW[l];
        k_cvt_feat<<<(ftot + 255) / 256, 256, 0, stream>>>(feat, fT, C[l], HW[l]);

        // one wave per 16x16 tile: (HW/16) M-tiles x 16 N-tiles
        k_gemm_head<<<(HW[l] / 16) * 16, 32, 0, stream>>>(fT, wcat, bcat, outg, C[l]);

        int stot = HW[l] * BPC * NC;
        unsigned int* hist = (unsigned int*)(ws + o_hist) + l * NBINS;
        k_hist<<<(stot + 255) / 256, 256, 0, stream>>>(outg, hist, HW[l]);

        float* thr = (float*)(ws + o_thr) + l;
        k_find_thr<<<1, 32, 0, stream>>>(hist, thr);

        int* cnt = (int*)(ws + o_cnt) + l;
        k_collect<<<(stot + 255) / 256, 256, 0, stream>>>(
            outg, thr, cnt,
            (float*)(ws + o_sc), (int*)(ws + o_lab), (float*)(ws + o_box),
            l * TOPK, HW[l], S[l], STR[l],
            ANC[l][0], ANC[l][1], ANC[l][2], ANC[l][3], ANC[l][4], ANC[l][5]);
    }

    k_rank<<<1, 1024, 0, stream>>>((float*)(ws + o_sc), (int*)(ws + o_ord));

    k_nms_out<<<1, 1024, 0, stream>>>(
        (float*)(ws + o_sc), (int*)(ws + o_lab), (float*)(ws + o_box),
        (int*)(ws + o_ord), (float*)d_out);
}
